// CrossAttentionLayer_33200097199043
// MI455X (gfx1250) — compile-verified, matrix operand order checked by reading, not run on hardware
//
#include <hip/hip_runtime.h>

typedef __attribute__((ext_vector_type(16))) _Float16 v16h;
typedef __attribute__((ext_vector_type(4)))  _Float16 v4h;
typedef __attribute__((ext_vector_type(8)))  float    v8f;
typedef __attribute__((ext_vector_type(4)))  unsigned int v4u;
typedef __attribute__((ext_vector_type(8)))  int      v8i;
typedef __attribute__((ext_vector_type(4)))  int      v4i;

#define B_  4
#define H_  8
#define D_  768
#define HD_ 96
#define LQ_ 1024
#define LK_ 4096

__device__ __forceinline__ v8f wmma_f16(v16h a, v16h b, v8f c) {
  // 8 args: (neg_a, A, neg_b, B, c_mod, C, reuse_a, reuse_b)
  return __builtin_amdgcn_wmma_f32_16x16x32_f16(false, a, false, b, (short)0, c,
                                                false, false);
}

// Reductions across the 16-lane halves of a wave32 (matches the 16x16 C/D
// layout where a matrix row lives across lanes 0-15 or 16-31).
__device__ __forceinline__ float half16_max(float v) {
#pragma unroll
  for (int m = 1; m < 16; m <<= 1) v = fmaxf(v, __shfl_xor(v, m, 32));
  return v;
}
__device__ __forceinline__ float half16_sum(float v) {
#pragma unroll
  for (int m = 1; m < 16; m <<= 1) v += __shfl_xor(v, m, 32);
  return v;
}

__global__ void f32_to_f16_kernel(const float* __restrict__ in,
                                  _Float16* __restrict__ out, int n4) {
  int i = blockIdx.x * blockDim.x + threadIdx.x;
  if (i < n4) {
    float4 f = ((const float4*)in)[i];
    v4h h = {(_Float16)f.x, (_Float16)f.y, (_Float16)f.z, (_Float16)f.w};
    ((v4h*)out)[i] = h;
  }
}

// ---------------------------------------------------------------------------
// Projection GEMM: out_headmajor[B,H,L,96] = X(M x 768) * W^T + bias
// Block tile 128x128, 8 waves, each wave owns a 16x128 strip.
// ---------------------------------------------------------------------------
__global__ __launch_bounds__(256) void proj_gemm_kernel(
    const _Float16* __restrict__ X, const _Float16* __restrict__ W,
    const float* __restrict__ bias, _Float16* __restrict__ out, int L) {
  __shared__ __align__(32) _Float16 At[128 * 32];
  __shared__ __align__(32) _Float16 Bt[128 * 32];
  const int tid  = threadIdx.x;
  const int wave = tid >> 5;
  const int lane = tid & 31;
  const int lm = lane & 15, lg = lane >> 4;
  const int m0 = blockIdx.x * 128;
  const int n0 = blockIdx.y * 128;

  const v8f vzero = {};
  v8f acc[8];
#pragma unroll
  for (int i = 0; i < 8; ++i) acc[i] = vzero;

  for (int k0 = 0; k0 < D_; k0 += 32) {
    // 128 rows x 4 uint4 per tile; 2 uniform stores per thread per tile
#pragma unroll
    for (int i = 0; i < 2; ++i) {
      int t = tid + i * 256;
      int r = t >> 2, c = t & 3;
      ((uint4*)At)[t] = *(const uint4*)(X + (size_t)(m0 + r) * D_ + k0 + c * 8);
      ((uint4*)Bt)[t] = *(const uint4*)(W + (size_t)(n0 + r) * D_ + k0 + c * 8);
    }
    __syncthreads();
    v16h a = *(const v16h*)&At[(wave * 16 + lm) * 32 + lg * 16];
    v16h bfrag[8];
#pragma unroll
    for (int nt = 0; nt < 8; ++nt)
      bfrag[nt] = *(const v16h*)&Bt[(nt * 16 + lm) * 32 + lg * 16];
#pragma unroll
    for (int nt = 0; nt < 8; ++nt)
      acc[nt] = wmma_f16(a, bfrag[nt], acc[nt]);
    __syncthreads();
  }

#pragma unroll
  for (int nt = 0; nt < 8; ++nt) {
    int n = n0 + nt * 16 + lm;
    int h = n / HD_, d = n % HD_;
    float bb = bias[n];
#pragma unroll
    for (int r = 0; r < 8; ++r) {
      int m = m0 + wave * 16 + r + 8 * lg;
      int bidx = m / L, l = m - bidx * L;
      out[(((size_t)bidx * H_ + h) * L + l) * HD_ + d] =
          (_Float16)(acc[nt][r] + bb);
    }
  }
}

// ---------------------------------------------------------------------------
// Output GEMM: d_out(M x 768 fp32) = A(M x 768 f16) * W^T + bias
// ---------------------------------------------------------------------------
__global__ __launch_bounds__(256) void out_gemm_kernel(
    const _Float16* __restrict__ A, const _Float16* __restrict__ W,
    const float* __restrict__ bias, float* __restrict__ out) {
  __shared__ __align__(32) _Float16 At[128 * 32];
  __shared__ __align__(32) _Float16 Bt[128 * 32];
  const int tid  = threadIdx.x;
  const int wave = tid >> 5;
  const int lane = tid & 31;
  const int lm = lane & 15, lg = lane >> 4;
  const int m0 = blockIdx.x * 128;
  const int n0 = blockIdx.y * 128;

  const v8f vzero = {};
  v8f acc[8];
#pragma unroll
  for (int i = 0; i < 8; ++i) acc[i] = vzero;

  for (int k0 = 0; k0 < D_; k0 += 32) {
#pragma unroll
    for (int i = 0; i < 2; ++i) {
      int t = tid + i * 256;
      int r = t >> 2, c = t & 3;
      ((uint4*)At)[t] = *(const uint4*)(A + (size_t)(m0 + r) * D_ + k0 + c * 8);
      ((uint4*)Bt)[t] = *(const uint4*)(W + (size_t)(n0 + r) * D_ + k0 + c * 8);
    }
    __syncthreads();
    v16h a = *(const v16h*)&At[(wave * 16 + lm) * 32 + lg * 16];
    v16h bfrag[8];
#pragma unroll
    for (int nt = 0; nt < 8; ++nt)
      bfrag[nt] = *(const v16h*)&Bt[(nt * 16 + lm) * 32 + lg * 16];
#pragma unroll
    for (int nt = 0; nt < 8; ++nt)
      acc[nt] = wmma_f16(a, bfrag[nt], acc[nt]);
    __syncthreads();
  }

#pragma unroll
  for (int nt = 0; nt < 8; ++nt) {
    int n = n0 + nt * 16 + lm;
    float bb = bias[n];
#pragma unroll
    for (int r = 0; r < 8; ++r) {
      int m = m0 + wave * 16 + r + 8 * lg;
      out[(size_t)m * D_ + n] = acc[nt][r] + bb;
    }
  }
}

// ---------------------------------------------------------------------------
// 3D rotary PE, in place on head-major [B,H,L,96] f16 data.
// ---------------------------------------------------------------------------
__global__ void rope_kernel(_Float16* __restrict__ X,
                            const float* __restrict__ coords, int L) {
  int idx = blockIdx.x * blockDim.x + threadIdx.x;
  int total = B_ * H_ * L * 48;
  if (idx >= total) return;
  int p = idx & 15;
  int a = (idx >> 4) % 3;
  int l = (idx / 48) % L;
  int h = (idx / (48 * L)) % H_;
  int b = idx / (48 * L * H_);
  float c = coords[((size_t)b * L + l) * 3 + a];
  float inv = __powf(10000.0f, -(float)p * (1.0f / 16.0f));
  float th = c * inv;
  float sn, cs;
  __sincosf(th, &sn, &cs);
  _Float16* base = X + (((size_t)b * H_ + h) * L + l) * HD_ + a * 32;
  float x1 = (float)base[p];
  float x2 = (float)base[16 + p];
  base[p]      = (_Float16)(x1 * cs - x2 * sn);
  base[16 + p] = (_Float16)(x1 * sn + x2 * cs);
}

// ---------------------------------------------------------------------------
// Flash attention: grid(8 q-tiles, H, B), 256 threads = 8 waves.
// K tiles are streamed into LDS by the Tensor Data Mover (TDM); V tiles are
// staged transposed by vector lanes; P re-layout goes through per-wave LDS.
// ---------------------------------------------------------------------------
__global__ __launch_bounds__(256) void flash_attn_kernel(
    const _Float16* __restrict__ Q, const _Float16* __restrict__ K,
    const _Float16* __restrict__ V, _Float16* __restrict__ O) {
  __shared__ __align__(32) _Float16 kt[64 * HD_];    // K tile, row-major
  __shared__ __align__(32) _Float16 vt[HD_ * 64];    // V tile, transposed
  __shared__ __align__(32) _Float16 pt[8 * 16 * 64]; // per-wave P staging
  const int tid  = threadIdx.x;
  const int wave = tid >> 5;
  const int lane = tid & 31;
  const int lm = lane & 15, lg = lane >> 4;
  const int b = blockIdx.z, h = blockIdx.y;
  const int q0 = blockIdx.x * 128 + wave * 16;

  const _Float16* qb = Q + (((size_t)b * H_ + h) * LQ_ + q0) * HD_;
  const _Float16* kb = K + ((size_t)b * H_ + h) * LK_ * HD_;
  const _Float16* vb = V + ((size_t)b * H_ + h) * LK_ * HD_;

  // Q fragments, resident for the whole kernel (16x96 = 3 K-steps of 32)
  v16h aq[3];
#pragma unroll
  for (int kk = 0; kk < 3; ++kk)
    aq[kk] = *(const v16h*)(qb + lm * HD_ + kk * 32 + lg * 16);

  const v8f vzero = {};
  v8f oacc[6];
#pragma unroll
  for (int i = 0; i < 6; ++i) oacc[i] = vzero;
  float mrun[8], lrun[8];
#pragma unroll
  for (int r = 0; r < 8; ++r) { mrun[r] = -1e30f; lrun[r] = 0.0f; }

  const float scale = 0.10206207262f;  // 1/sqrt(96)
  const unsigned int kt_lds = (unsigned int)(size_t)(&kt[0]);

  for (int c0 = 0; c0 < LK_; c0 += 64) {
    // --- V tile: staged transposed (24 uniform scalar moves per thread) ---
#pragma unroll
    for (int i = 0; i < 24; ++i) {
      int t = tid + i * 256;
      int r = t / HD_, d = t - r * HD_;
      vt[d * 64 + r] = vb[(size_t)(c0 + r) * HD_ + d];
    }
    if (c0 + 64 < LK_)
      __builtin_prefetch(vb + (size_t)(c0 + 64) * HD_ + tid * 24, 0, 1);

    // --- K tile: Tensor Data Mover, issued by wave 0 only -----------------
#if __has_builtin(__builtin_amdgcn_tensor_load_to_lds)
    if (wave == 0) {
      unsigned long long ga =
          (unsigned long long)(size_t)(kb + (size_t)c0 * HD_);
      // D# group0: count=1 | lds_addr | 57-bit global addr | type=2
      v4u g0 = {1u, kt_lds, (unsigned int)(ga & 0xffffffffu),
                (unsigned int)((ga >> 32) & 0x1ffffffu) | 0x80000000u};
      // D# group1: data_size=2B; tensor_dim0=96 (stride 96), tensor_dim1=4096;
      //            tile 96 x 64
      v8i g1 = {(int)0x00010000,       // wg_mask=0, data_size=1 (2 bytes)
                (int)(96u << 16),      // tensor_dim0[15:0] (=96) in [31:16]
                (int)(4096u << 16),    // tensor_dim0 hi=0, tensor_dim1 lo=4096
                (int)(96u << 16),      // tensor_dim1 hi=0, tile_dim0=96
                (int)64,               // tile_dim1=64, tile_dim2=0
                (int)96,               // tensor_dim0_stride lo = 96
                0, 0};                 // stride hi, tensor_dim1_stride
      v4i gz4 = {0, 0, 0, 0};
      v8i gz8 = {0, 0, 0, 0, 0, 0, 0, 0};
      // 6-arg form (clang-23 / therock-10.0 headers)
      __builtin_amdgcn_tensor_load_to_lds(g0, g1, gz4, gz4, gz8, 0);
      __builtin_amdgcn_s_wait_tensorcnt(0);
    }
#else
    for (int i = 0; i < 3; ++i) {
      int t = tid + i * 256;  // 64*96 halfs = 768 uint4
      ((uint4*)kt)[t] = ((const uint4*)(kb + (size_t)c0 * HD_))[t];
    }
#endif
    __syncthreads();

    // --- S = Q * K^T  (4 N-tiles x 3 K-steps) -----------------------------
    v8f s[4];
#pragma unroll
    for (int i = 0; i < 4; ++i) s[i] = vzero;
#pragma unroll
    for (int kk = 0; kk < 3; ++kk) {
      v16h bk[4];
#pragma unroll
      for (int nt = 0; nt < 4; ++nt)
        bk[nt] = *(const v16h*)&kt[(nt * 16 + lm) * HD_ + kk * 32 + lg * 16];
#pragma unroll
      for (int nt = 0; nt < 4; ++nt)
        s[nt] = wmma_f16(aq[kk], bk[nt], s[nt]);
    }

    // --- online softmax ---------------------------------------------------
    float rmax[8];
#pragma unroll
    for (int r = 0; r < 8; ++r) {
      float v = s[0][r] * scale;
      v = fmaxf(v, s[1][r] * scale);
      v = fmaxf(v, s[2][r] * scale);
      v = fmaxf(v, s[3][r] * scale);
      rmax[r] = half16_max(v);
    }
    float alpha[8], psum[8];
#pragma unroll
    for (int r = 0; r < 8; ++r) {
      float mnew = fmaxf(mrun[r], rmax[r]);
      alpha[r] = __expf(mrun[r] - mnew);
      mrun[r] = mnew;
      psum[r] = 0.0f;
    }
#pragma unroll
    for (int nt = 0; nt < 4; ++nt) {
#pragma unroll
      for (int r = 0; r < 8; ++r) {
        float pv = __expf(s[nt][r] * scale - mrun[r]);
        psum[r] += pv;
        // re-layout D-format (row r+8*lg across lanes) -> A-format via LDS
        pt[wave * 1024 + (r + 8 * lg) * 64 + nt * 16 + lm] = (_Float16)pv;
      }
    }
#pragma unroll
    for (int r = 0; r < 8; ++r)
      lrun[r] = lrun[r] * alpha[r] + half16_sum(psum[r]);
#pragma unroll
    for (int nt = 0; nt < 6; ++nt)
#pragma unroll
      for (int r = 0; r < 8; ++r) oacc[nt][r] *= alpha[r];

    // wave-internal cross-lane traffic through LDS: wait on DS counter only
    asm volatile("s_wait_dscnt 0" ::: "memory");

    // --- O += P * V  (2 K-steps x 6 N-tiles) ------------------------------
#pragma unroll
    for (int ks = 0; ks < 2; ++ks) {
      v16h ap = *(const v16h*)&pt[wave * 1024 + lm * 64 + ks * 32 + lg * 16];
      v16h bv[6];
#pragma unroll
      for (int nt = 0; nt < 6; ++nt)
        bv[nt] = *(const v16h*)&vt[(nt * 16 + lm) * 64 + ks * 32 + lg * 16];
#pragma unroll
      for (int nt = 0; nt < 6; ++nt)
        oacc[nt] = wmma_f16(ap, bv[nt], oacc[nt]);
    }
    __syncthreads();
  }

  // normalize and write attention output in [B, Lq, H*96] layout (f16)
#pragma unroll
  for (int nt = 0; nt < 6; ++nt) {
#pragma unroll
    for (int r = 0; r < 8; ++r) {
      int m = q0 + r + 8 * lg;
      float o = oacc[nt][r] / lrun[r];
      O[((size_t)b * LQ_ + m) * D_ + h * HD_ + nt * 16 + lm] = (_Float16)o;
    }
  }
}

// ---------------------------------------------------------------------------
extern "C" void kernel_launch(void* const* d_in, const int* in_sizes, int n_in,
                              void* d_out, int out_size, void* d_ws,
                              size_t ws_size, hipStream_t stream) {
  const float* query = (const float*)d_in[0];
  const float* key_  = (const float*)d_in[1];
  const float* value = (const float*)d_in[2];
  const float* cq    = (const float*)d_in[3];
  const float* ck    = (const float*)d_in[4];
  const float* Wq    = (const float*)d_in[5];
  const float* bq    = (const float*)d_in[6];
  const float* Wk    = (const float*)d_in[7];
  const float* bk    = (const float*)d_in[8];
  const float* Wv    = (const float*)d_in[9];
  const float* bv    = (const float*)d_in[10];
  const float* Wo    = (const float*)d_in[11];
  const float* bo    = (const float*)d_in[12];

  _Float16* p = (_Float16*)d_ws;
  _Float16* qh  = p; p += (size_t)4096 * 768;
  _Float16* kh  = p; p += (size_t)16384 * 768;
  _Float16* vh  = p; p += (size_t)16384 * 768;
  _Float16* Wqh = p; p += (size_t)768 * 768;
  _Float16* Wkh = p; p += (size_t)768 * 768;
  _Float16* Wvh = p; p += (size_t)768 * 768;
  _Float16* Woh = p; p += (size_t)768 * 768;
  _Float16* Qp  = p; p += (size_t)4096 * 768;
  _Float16* Kp  = p; p += (size_t)16384 * 768;
  _Float16* Vp  = p; p += (size_t)16384 * 768;
  _Float16* Ao  = p; p += (size_t)4096 * 768;

  auto cvt = [&](const float* s, _Float16* d, int n) {
    int n4 = n / 4;
    f32_to_f16_kernel<<<dim3((n4 + 255) / 256), dim3(256), 0, stream>>>(s, d,
                                                                        n4);
  };
  cvt(query, qh, 4096 * 768);
  cvt(key_,  kh, 16384 * 768);
  cvt(value, vh, 16384 * 768);
  cvt(Wq, Wqh, 768 * 768);
  cvt(Wk, Wkh, 768 * 768);
  cvt(Wv, Wvh, 768 * 768);
  cvt(Wo, Woh, 768 * 768);

  proj_gemm_kernel<<<dim3(32, 6),  dim3(256), 0, stream>>>(qh, Wqh, bq, Qp, LQ_);
  proj_gemm_kernel<<<dim3(128, 6), dim3(256), 0, stream>>>(kh, Wkh, bk, Kp, LK_);
  proj_gemm_kernel<<<dim3(128, 6), dim3(256), 0, stream>>>(vh, Wvh, bv, Vp, LK_);

  rope_kernel<<<dim3((B_ * H_ * LQ_ * 48 + 255) / 256), dim3(256), 0, stream>>>(
      Qp, cq, LQ_);
  rope_kernel<<<dim3((B_ * H_ * LK_ * 48 + 255) / 256), dim3(256), 0, stream>>>(
      Kp, ck, LK_);

  flash_attn_kernel<<<dim3(8, H_, B_), dim3(256), 0, stream>>>(Qp, Kp, Vp, Ao);

  out_gemm_kernel<<<dim3(32, 6), dim3(256), 0, stream>>>(Ao, Woh, bo,
                                                         (float*)d_out);
  (void)in_sizes; (void)n_in; (void)out_size; (void)ws_size;
}
